// RNNModule_6597069766954
// MI455X (gfx1250) — matrix operand order repaired and checked by reading, not compiled, not run on hardware
//
#include <hip/hip_runtime.h>
#include <hip/hip_bf16.h>

typedef __bf16 bf16;
typedef __attribute__((ext_vector_type(16))) __bf16 v16bf;
typedef __attribute__((ext_vector_type(8)))  float  v8f;

#define VOCAB  32000
#define EMBED  512
#define HIDDEN 1024
#define NBATCH 16
#define SEQ    256
#define PADIDX 3
#define MTOT   (NBATCH * SEQ)   // 4096

// ---------------- kernel 0: fp32 -> bf16 bulk convert (8 elems/thread) ----------------
__global__ __launch_bounds__(256) void cvt_f32_bf16(const float* __restrict__ src,
                                                    bf16* __restrict__ dst, int n8) {
  int i = blockIdx.x * blockDim.x + threadIdx.x;
  if (i >= n8) return;
  const float4* s = (const float4*)src + (size_t)i * 2;
  float4 a = s[0], b = s[1];
  union { bf16 h[8]; uint4 u; } cv;
  cv.h[0] = (bf16)a.x; cv.h[1] = (bf16)a.y; cv.h[2] = (bf16)a.z; cv.h[3] = (bf16)a.w;
  cv.h[4] = (bf16)b.x; cv.h[5] = (bf16)b.y; cv.h[6] = (bf16)b.z; cv.h[7] = (bf16)b.w;
  ((uint4*)dst)[i] = cv.u;
}

// ---------------- kernel 1: embedding gather + input projection ----------------
// xproj[m, n] = sum_e emb[ids[m], e] * Wxh[n, e] + bias[n], m = b*SEQ + s
// (fp32 sources need on-the-fly bf16 conversion, so this one stages via VGPRs.)
__global__ __launch_bounds__(256)
void embed_xproj(const int* __restrict__ ids, const float* __restrict__ emb,
                 const float* __restrict__ Wxh, const float* __restrict__ bias,
                 float* __restrict__ xproj) {
  __shared__ __align__(16) bf16 As[128][40];  // [m][k], 8-elem pad (80B row stride)
  __shared__ __align__(16) bf16 Bs[128][40];  // [n][k]
  const int tid   = threadIdx.x;
  const int lane  = tid & 31;
  const int w     = tid >> 5;       // 0..7
  const int wm    = w >> 2;         // 0..1
  const int wn    = w & 3;          // 0..3
  const int lhalf = (lane >> 4) & 1;
  const int l15   = lane & 15;
  const int mblk  = blockIdx.y * 128;
  const int nblk  = blockIdx.x * 128;

  v8f acc[4][2] = {};

  for (int kb = 0; kb < EMBED; kb += 32) {
    #pragma unroll
    for (int i = 0; i < 4; ++i) {               // 1024 float4 tasks / 256 threads
      int task = tid + i * 256;
      int r = task >> 3;
      int c = (task & 7) * 4;
      int id = ids[mblk + r];
      float4 va = make_float4(0.f, 0.f, 0.f, 0.f);
      if (id != PADIDX) va = *(const float4*)(emb + (size_t)id * EMBED + kb + c);
      union { bf16 h[4]; uint2 u; } ca, cb;
      ca.h[0] = (bf16)va.x; ca.h[1] = (bf16)va.y; ca.h[2] = (bf16)va.z; ca.h[3] = (bf16)va.w;
      *(uint2*)&As[r][c] = ca.u;
      float4 vb = *(const float4*)(Wxh + (size_t)(nblk + r) * EMBED + kb + c);
      cb.h[0] = (bf16)vb.x; cb.h[1] = (bf16)vb.y; cb.h[2] = (bf16)vb.z; cb.h[3] = (bf16)vb.w;
      *(uint2*)&Bs[r][c] = cb.u;
    }
    __syncthreads();
    #pragma unroll
    for (int ms = 0; ms < 4; ++ms) {
      v16bf a;
      const bf16* ap = &As[wm * 64 + ms * 16 + l15][lhalf * 8];
      *(uint4*)&a       = *(const uint4*)ap;
      *((uint4*)&a + 1) = *(const uint4*)(ap + 16);
      #pragma unroll
      for (int ns = 0; ns < 2; ++ns) {
        v16bf b;
        const bf16* bp = &Bs[wn * 32 + ns * 16 + l15][lhalf * 8];
        *(uint4*)&b       = *(const uint4*)bp;
        *((uint4*)&b + 1) = *(const uint4*)(bp + 16);
        acc[ms][ns] = __builtin_amdgcn_wmma_f32_16x16x32_bf16(
            false, a, false, b, (short)0, acc[ms][ns], false, false);
      }
    }
    __syncthreads();
  }
  #pragma unroll
  for (int ms = 0; ms < 4; ++ms)
    #pragma unroll
    for (int ns = 0; ns < 2; ++ns) {
      int n = nblk + wn * 32 + ns * 16 + l15;
      float bz = bias[n];
      #pragma unroll
      for (int r = 0; r < 8; ++r) {
        int m = mblk + wm * 64 + ms * 16 + r + lhalf * 8;
        xproj[(size_t)m * HIDDEN + n] = acc[ms][ns][r] + bz;
      }
    }
}

// ---------------- kernel 2: one recurrence step (split-K over 4 waves) ----------------
// h_out = tanh(xproj[:, s, :] + h_in @ Whh^T). Block = 128 thr (4 waves) owns one
// 16x16 N-tile; wave w accumulates K-chunk [w*256, w*256+256) in 8 WMMA steps,
// partials reduced through LDS. 64 blocks -> short serial chain per step.
__global__ __launch_bounds__(128)
void rnn_step(const float* __restrict__ h_in, const bf16* __restrict__ whh,
              const float* __restrict__ xproj, float* __restrict__ h_out,
              bf16* __restrict__ hall, int s) {
  __shared__ float red[4][8][32];
  const int tid   = threadIdx.x;
  const int lane  = tid & 31;
  const int w     = tid >> 5;                   // K-chunk 0..3
  const int lhalf = (lane >> 4) & 1;
  const int l15   = lane & 15;
  const int n0    = blockIdx.x * 16;
  const int k0    = w * (HIDDEN / 4);

  v8f acc = {};
  #pragma unroll
  for (int ki = 0; ki < 8; ++ki) {
    const int kb = k0 + ki * 32;
    v16bf a;                                    // convert fp32 h on the fly
    const float* hr = h_in + (size_t)l15 * HIDDEN + kb + lhalf * 8;
    #pragma unroll
    for (int j = 0; j < 8; ++j) a[j] = (bf16)hr[j];
    #pragma unroll
    for (int j = 0; j < 8; ++j) a[8 + j] = (bf16)hr[16 + j];
    v16bf b;
    const bf16* wr = whh + (size_t)(n0 + l15) * HIDDEN + kb + lhalf * 8;
    *(uint4*)&b       = *(const uint4*)wr;
    *((uint4*)&b + 1) = *(const uint4*)(wr + 16);
    acc = __builtin_amdgcn_wmma_f32_16x16x32_bf16(false, a, false, b, (short)0, acc,
                                                  false, false);
  }
  #pragma unroll
  for (int r = 0; r < 8; ++r) red[w][r][lane] = acc[r];
  __syncthreads();
  if (w == 0) {
    #pragma unroll
    for (int r = 0; r < 8; ++r) {
      float v = red[0][r][lane] + red[1][r][lane] + red[2][r][lane] + red[3][r][lane];
      int m = r + lhalf * 8;                    // batch index
      int n = n0 + l15;
      v = tanhf(v + xproj[((size_t)m * SEQ + s) * HIDDEN + n]);
      h_out[(size_t)m * HIDDEN + n] = v;
      hall[((size_t)s * NBATCH + m) * HIDDEN + n] = (bf16)v;   // row = s*16 + b
    }
  }
}

// ---------------- kernel 3: logits = h_all @ Who^T + b ----------------
// M=4096, N=32000, K=1024. 128x256 block tile (8 waves, 64x64 each), K-step 32.
// Both operands are bf16 in memory -> tiles are DMA'd into double-buffered LDS with
// GLOBAL_LOAD_ASYNC_TO_LDS_B128 (ASYNCcnt), overlapping copy with WMMA.
__global__ __launch_bounds__(256)
void logits_gemm(const bf16* __restrict__ hall, const bf16* __restrict__ who,
                 const float* __restrict__ bias, float* __restrict__ out) {
  __shared__ __align__(16) bf16 As[2][128][40];   // 2 x 10240 B
  __shared__ __align__(16) bf16 Bs[2][256][40];   // 2 x 20480 B
  const int tid   = threadIdx.x;
  const int lane  = tid & 31;
  const int w     = tid >> 5;       // 0..7
  const int wm    = w >> 2;         // 0..1  (64-row half)
  const int wn    = w & 3;          // 0..3  (64-col quarter)
  const int lhalf = (lane >> 4) & 1;
  const int l15   = lane & 15;
  const int mblk  = blockIdx.y * 128;
  const int nblk  = blockIdx.x * 256;

  // per-thread async-copy tasks: A = 512 b128 chunks, B = 1024 b128 chunks
  unsigned long long gaA[2], gaB[4];
  unsigned ldsA[2], ldsB[4];
  #pragma unroll
  for (int i = 0; i < 2; ++i) {
    int task = tid + i * 256;
    int r = task >> 2, c = (task & 3) * 8;
    gaA[i]  = (unsigned long long)(hall + (size_t)(mblk + r) * HIDDEN + c);
    ldsA[i] = (unsigned)(unsigned long long)&As[0][r][c];
  }
  #pragma unroll
  for (int i = 0; i < 4; ++i) {
    int task = tid + i * 256;
    int r = task >> 2, c = (task & 3) * 8;
    gaB[i]  = (unsigned long long)(who + (size_t)(nblk + r) * HIDDEN + c);
    ldsB[i] = (unsigned)(unsigned long long)&Bs[0][r][c];
  }

  auto issue_tile = [&](int kb, int buf) {
    #pragma unroll
    for (int i = 0; i < 2; ++i) {
      unsigned lo = ldsA[i] + (unsigned)buf * 10240u;
      unsigned long long ga = gaA[i] + (unsigned long long)kb * 2ull;
      asm volatile("global_load_async_to_lds_b128 %0, %1, off"
                   :: "v"(lo), "v"(ga) : "memory");
    }
    #pragma unroll
    for (int i = 0; i < 4; ++i) {
      unsigned lo = ldsB[i] + (unsigned)buf * 20480u;
      unsigned long long ga = gaB[i] + (unsigned long long)kb * 2ull;
      asm volatile("global_load_async_to_lds_b128 %0, %1, off"
                   :: "v"(lo), "v"(ga) : "memory");
    }
  };

  v8f acc[4][4] = {};

  issue_tile(0, 0);
  for (int kt = 0; kt < HIDDEN / 32; ++kt) {
    const int buf = kt & 1;
    if (kt + 1 < HIDDEN / 32) {
      issue_tile((kt + 1) * 32, buf ^ 1);
      asm volatile("s_wait_asynccnt 0x6" ::: "memory");   // tile kt done; 6 newer in flight
    } else {
      asm volatile("s_wait_asynccnt 0x0" ::: "memory");
    }
    __syncthreads();                                      // tile kt visible to all waves
    #pragma unroll
    for (int ms = 0; ms < 4; ++ms) {
      v16bf a;
      const bf16* ap = &As[buf][wm * 64 + ms * 16 + l15][lhalf * 8];
      *(uint4*)&a       = *(const uint4*)ap;
      *((uint4*)&a + 1) = *(const uint4*)(ap + 16);
      #pragma unroll
      for (int ns = 0; ns < 4; ++ns) {
        v16bf b;
        const bf16* bp = &Bs[buf][wn * 64 + ns * 16 + l15][lhalf * 8];
        *(uint4*)&b       = *(const uint4*)bp;
        *((uint4*)&b + 1) = *(const uint4*)(bp + 16);
        acc[ms][ns] = __builtin_amdgcn_wmma_f32_16x16x32_bf16(
            false, a, false, b, (short)0, acc[ms][ns], false, false);
      }
    }
    __syncthreads();                                      // done reading buf before reuse
  }

  #pragma unroll
  for (int ms = 0; ms < 4; ++ms)
    #pragma unroll
    for (int ns = 0; ns < 4; ++ns) {
      int n = nblk + wn * 64 + ns * 16 + l15;
      float bz = bias[n];
      #pragma unroll
      for (int r = 0; r < 8; ++r) {
        int g = mblk + wm * 64 + ms * 16 + r + lhalf * 8;   // g = s*16 + b
        int ss = g >> 4, bb = g & 15;
        out[((size_t)bb * SEQ + ss) * VOCAB + n] = acc[ms][ns][r] + bz;
      }
    }
}

// ---------------- host orchestration ----------------
extern "C" void kernel_launch(void* const* d_in, const int* in_sizes, int n_in,
                              void* d_out, int out_size, void* d_ws, size_t ws_size,
                              hipStream_t stream) {
  const int*   ids     = (const int*)d_in[0];
  const float* hidden0 = (const float*)d_in[1];
  const float* emb     = (const float*)d_in[2];
  const float* Wxh_w   = (const float*)d_in[3];
  const float* Wxh_b   = (const float*)d_in[4];
  const float* Whh_w   = (const float*)d_in[5];
  const float* Who_w   = (const float*)d_in[6];
  const float* Who_b   = (const float*)d_in[7];
  float* out = (float*)d_out;

  char* ws = (char*)d_ws;
  bf16*  who_bf = (bf16*)ws;  ws += (size_t)VOCAB  * HIDDEN * sizeof(bf16);   // 65.5 MB
  bf16*  whh_bf = (bf16*)ws;  ws += (size_t)HIDDEN * HIDDEN * sizeof(bf16);   //  2.1 MB
  float* xproj  = (float*)ws; ws += (size_t)MTOT   * HIDDEN * sizeof(float);  // 16.8 MB
  bf16*  hall   = (bf16*)ws;  ws += (size_t)MTOT   * HIDDEN * sizeof(bf16);   //  8.4 MB
  float* hb0    = (float*)ws; ws += (size_t)NBATCH * HIDDEN * sizeof(float);
  float* hb1    = (float*)ws;

  { int n8 = VOCAB * HIDDEN / 8;
    cvt_f32_bf16<<<(n8 + 255) / 256, 256, 0, stream>>>(Who_w, who_bf, n8); }
  { int n8 = HIDDEN * HIDDEN / 8;
    cvt_f32_bf16<<<(n8 + 255) / 256, 256, 0, stream>>>(Whh_w, whh_bf, n8); }

  embed_xproj<<<dim3(HIDDEN / 128, MTOT / 128), 256, 0, stream>>>(ids, emb, Wxh_w,
                                                                  Wxh_b, xproj);
  const float* hprev = hidden0;
  for (int s = 0; s < SEQ; ++s) {
    float* hcur = (s & 1) ? hb1 : hb0;
    rnn_step<<<dim3(HIDDEN / 16), 128, 0, stream>>>(hprev, whh_bf, xproj, hcur, hall, s);
    hprev = hcur;
  }

  logits_gemm<<<dim3(VOCAB / 256, MTOT / 128), 256, 0, stream>>>(hall, who_bf, Who_b, out);

  hipMemcpyAsync(out + (size_t)NBATCH * SEQ * VOCAB, hprev,
                 (size_t)NBATCH * HIDDEN * sizeof(float),
                 hipMemcpyDeviceToDevice, stream);
}
